// BAmutil_86852828660054
// MI455X (gfx1250) — compile-verified, matrix operand order checked by reading, not run on hardware
//
#include <hip/hip_runtime.h>
#include <hip/hip_bf16.h>

// ---------------------------------------------------------------------------
// CDNA5 (gfx1250) windowed-attention pipeline.
// GEMMs: v_wmma_f32_16x16x32_bf16 with LDS tiles pre-swizzled into per-lane
// fragment order (fragment load = 2x ds_load_b128). Staging uses async
// global->LDS (ASYNCcnt) when the toolchain exposes the gfx1250 builtins.
// ---------------------------------------------------------------------------

typedef __bf16 bhalf;
typedef __attribute__((ext_vector_type(16))) __bf16 v16bf;
typedef __attribute__((ext_vector_type(8)))  float  v8f;

#define WMMA_BF16(a, b, c) \
  __builtin_amdgcn_wmma_f32_16x16x32_bf16(false, (a), false, (b), (short)0, (c), false, false)

#if __has_builtin(__builtin_amdgcn_global_load_async_to_lds_b32) && \
    __has_builtin(__builtin_amdgcn_s_wait_asynccnt)
#define HAVE_ASYNC_LDS 1
#else
#define HAVE_ASYNC_LDS 0
#endif

#if HAVE_ASYNC_LDS
typedef __attribute__((address_space(1))) int gas_int;  // global
typedef __attribute__((address_space(3))) int las_int;  // LDS
#endif

// Copy one dword global->LDS. Async path: fire-and-forget DMA tracked by
// ASYNCcnt (no VGPR staging); fallback: plain load+store.
__device__ __forceinline__ void stageU32(const unsigned* g, unsigned* l) {
#if HAVE_ASYNC_LDS
  __builtin_amdgcn_global_load_async_to_lds_b32((gas_int*)g, (las_int*)l, 0, 0);
#else
  *l = *g;
#endif
}
__device__ __forceinline__ void asyncFence() {
#if HAVE_ASYNC_LDS
  __builtin_amdgcn_s_wait_asynccnt(0);
#endif
}

__device__ __forceinline__ v8f zero8() {
  v8f z;
#pragma unroll
  for (int i = 0; i < 8; ++i) z[i] = 0.f;
  return z;
}

// Offset (elements) of element (mn, k) inside a 512-element fragment tile in
// lane-major order: lane = mn + 16*((k>>3)&1), slot = (k&7)|((k>>4)<<3).
// Valid for both A (mn=M) and B (mn=N) 16-bit fragments (ISA 7.12.2).
__device__ __forceinline__ int fragOff(int mn, int k) {
  return ((mn + (((k >> 3) & 1) << 4)) << 4) + ((k & 7) | ((k >> 4) << 3));
}

// Load a pre-swizzled fragment: 32 contiguous bytes per lane (2x b128).
__device__ __forceinline__ v16bf loadFrag(const bhalf* tile) {
  return *(const v16bf*)(tile + ((threadIdx.x & 31) << 4));
}

// Pack two f32 -> bf16 pair dword (K1 staging: source is f32).
__device__ __forceinline__ unsigned pack2(float a, float b) {
  union { bhalf h[2]; unsigned u; } u;
  u.h[0] = (bhalf)a;
  u.h[1] = (bhalf)b;
  return u.u;
}
// Pack two bf16 -> dword (no conversions).
__device__ __forceinline__ unsigned pack2bf(bhalf a, bhalf b) {
  union { bhalf h[2]; unsigned u; } u;
  u.h[0] = a;
  u.h[1] = b;
  return u.u;
}
__device__ __forceinline__ void storePair(bhalf* tile, int off, unsigned p) {
  *(unsigned*)(tile + off) = p;  // off even by construction
}

// ---------------------------------------------------------------------------
// K1: QKV projection. Block per (window l, batch b). A = xw[64x128] gathered
// from the 8x8 window; B = W^T in 3 sections of 128 cols (q|k|v). 8 waves,
// one 16-col strip each; K loop 4x32. Outputs bf16 [b][h][L][t][cph].
// ---------------------------------------------------------------------------
__global__ __launch_bounds__(256) void qkv_kernel(
    const float* __restrict__ x, const float* __restrict__ W,
    const float* __restrict__ bias,
    bhalf* __restrict__ q, bhalf* __restrict__ k, bhalf* __restrict__ v) {
  __shared__ __align__(32) bhalf sA[16 * 512];   // A tiles [mt(4)][kk(4)]
  __shared__ __align__(32) bhalf sWB[32 * 512];  // B tiles [kk(4)][nt(8)]

  const int l = blockIdx.x;
  const int b = blockIdx.y;
  const int tid = threadIdx.x;
  const int wv = tid >> 5;
  const int wi = l >> 5, wj = l & 31;

  // Stage A: element (tt, ch) -> tile (tt>>4)*4 + (ch>>5), pairs along ch.
  {
    const float* xb = x + (((size_t)b * 128) * 256) * 256;
#pragma unroll 4
    for (int e = tid; e < 4096; e += 256) {
      int tt = e & 63;  // consecutive lanes -> consecutive tt (8-elem runs)
      int chp = e >> 6;
      int ch = chp * 2;
      int y = wi * 8 + (tt >> 3);
      int xc = wj * 8 + (tt & 7);
      size_t gi = (size_t)ch * 65536 + (size_t)y * 256 + xc;
      storePair(sA + ((tt >> 4) * 4 + (ch >> 5)) * 512, fragOff(tt & 15, ch & 31),
                pack2(xb[gi], xb[gi + 65536]));
    }
  }

  bhalf* const dsts[3] = {q, k, v};
  for (int sec = 0; sec < 3; ++sec) {
    __syncthreads();  // waves done reading previous section's B tiles
    // Stage B tiles: B[k][n] = W[sec*128+n][k]; tile (k>>5)*8 + (n>>4).
#pragma unroll 4
    for (int e = tid; e < 8192; e += 256) {
      int kp = e & 63;  // consecutive lanes -> consecutive k (coalesced W)
      int nn = e >> 6;
      int kk = kp * 2;
      const float* wr = W + (size_t)(sec * 128 + nn) * 128 + kk;
      storePair(sWB + ((kk >> 5) * 8 + (nn >> 4)) * 512, fragOff(nn & 15, kk & 31),
                pack2(wr[0], wr[1]));
    }
    __syncthreads();

    v8f acc[4];
#pragma unroll
    for (int mt = 0; mt < 4; ++mt) acc[mt] = zero8();
#pragma unroll
    for (int kk = 0; kk < 4; ++kk) {
      v16bf bm = loadFrag(sWB + (kk * 8 + wv) * 512);
#pragma unroll
      for (int mt = 0; mt < 4; ++mt) {
        v16bf am = loadFrag(sA + (mt * 4 + kk) * 512);
        acc[mt] = WMMA_BF16(am, bm, acc[mt]);
      }
    }

    // Epilogue: bias add, scatter into head-split layout.
    int ln = tid & 31;
    int n = ln & 15, mb = (ln >> 4) << 3;
    int cc128 = wv * 16 + n;
    int h = cc128 >> 5, cc = cc128 & 31;
    float bs = bias[sec * 128 + cc128];
    bhalf* dst = dsts[sec];
    size_t base = ((((size_t)b * 4 + h) * 1024 + l) * 64) * 32 + cc;
#pragma unroll
    for (int mt = 0; mt < 4; ++mt) {
#pragma unroll
      for (int r = 0; r < 8; ++r) {
        int tt = mt * 16 + mb + r;
        dst[base + (size_t)tt * 32] = (bhalf)(acc[mt][r] + bs);
      }
    }
  }
}

// ---------------------------------------------------------------------------
// K2: q_r/k_r = relu(mean over t). Block per (l, bh); wave0 -> q, wave1 -> k.
// ---------------------------------------------------------------------------
__global__ __launch_bounds__(64) void mean_kernel(
    const bhalf* __restrict__ q, const bhalf* __restrict__ k,
    bhalf* __restrict__ qr, bhalf* __restrict__ kr) {
  int l = blockIdx.x, bh = blockIdx.y;
  int wv = threadIdx.x >> 5, cc = threadIdx.x & 31;
  const bhalf* src = wv ? k : q;
  bhalf* dst = wv ? kr : qr;
  size_t base = (((size_t)bh * 1024) + l) * 2048 + cc;
  float s = 0.f;
#pragma unroll 8
  for (int tt = 0; tt < 64; ++tt) s += (float)src[base + (size_t)tt * 32];
  float m = s * (1.0f / 64.0f);
  dst[((size_t)bh * 1024 + l) * 32 + cc] = (bhalf)fmaxf(m, 0.f);
}

// ---------------------------------------------------------------------------
// K3: a_r = relu(q_r) @ relu(k_r)^T per (b,h). Block: 64x64 tile, K=32 in one
// WMMA step. 8 waves x 2 tiles. Staging = raw dword copies (async capable).
// ---------------------------------------------------------------------------
__global__ __launch_bounds__(256) void ar_kernel(
    const bhalf* __restrict__ qr, const bhalf* __restrict__ kr,
    bhalf* __restrict__ ar) {
  __shared__ __align__(32) bhalf sQ[4 * 512];  // A tiles mt=0..3
  __shared__ __align__(32) bhalf sK[4 * 512];  // B tiles nt=0..3 (B[k][n]=kr[n][k])
  int n0 = blockIdx.x * 64, m0 = blockIdx.y * 64, bh = blockIdx.z;
  int tid = threadIdx.x, wv = tid >> 5;
  size_t rbase = (size_t)bh * 1024 * 32;
#pragma unroll 2
  for (int e = tid; e < 1024; e += 256) {
    int row = e >> 4, ccp = e & 15;
    int off = fragOff(row & 15, ccp * 2);
    stageU32((const unsigned*)(qr + rbase + (size_t)(m0 + row) * 32) + ccp,
             (unsigned*)(sQ + (row >> 4) * 512 + off));
    stageU32((const unsigned*)(kr + rbase + (size_t)(n0 + row) * 32) + ccp,
             (unsigned*)(sK + (row >> 4) * 512 + off));
  }
  asyncFence();
  __syncthreads();
  size_t abase = (size_t)bh * 1024 * 1024;
  int ln = tid & 31, n = ln & 15, mb = (ln >> 4) << 3;
#pragma unroll
  for (int s = 0; s < 2; ++s) {
    int tIdx = wv * 2 + s;
    int mt = tIdx >> 2, nt = tIdx & 3;
    v16bf am = loadFrag(sQ + mt * 512);
    v16bf bm = loadFrag(sK + nt * 512);
    v8f acc = zero8();
    acc = WMMA_BF16(am, bm, acc);
#pragma unroll
    for (int r = 0; r < 8; ++r)
      ar[abase + (size_t)(m0 + mt * 16 + mb + r) * 1024 + (n0 + nt * 16 + n)] =
          (bhalf)acc[r];
  }
}

// ---------------------------------------------------------------------------
// K4 (dominant GEMM): dst = relu(a_r @ src), src viewed [1024 x 2048] per
// (b,h). Block tile 64x256, K loop 32x32, double-buffered LDS: stage slab
// j0+32 (async) while computing slab j0. Wave owns two 16-col strips.
// ---------------------------------------------------------------------------
__global__ __launch_bounds__(256) void apply_ar_kernel(
    const bhalf* __restrict__ ar, const bhalf* __restrict__ src,
    bhalf* __restrict__ dst) {
  __shared__ __align__(32) bhalf sA[2][4 * 512];   // A tiles mt=0..3
  __shared__ __align__(32) bhalf sB[2][16 * 512];  // B tiles nt=0..15
  int d0 = blockIdx.x * 256, m0 = blockIdx.y * 64, bh = blockIdx.z;
  int tid = threadIdx.x, wv = tid >> 5;
  size_t arb = (size_t)bh * 1024 * 1024;
  size_t srb = (size_t)bh * 1024 * 2048;

  v8f acc[4][2];
#pragma unroll
  for (int mt = 0; mt < 4; ++mt) {
    acc[mt][0] = zero8();
    acc[mt][1] = zero8();
  }

  auto stage = [&](int buf, int j0) {
    // A tiles: contiguous bf16 pairs -> raw dword copy (async if available).
#pragma unroll 2
    for (int e = tid; e < 1024; e += 256) {
      int row = e >> 4, jp = e & 15;
      stageU32((const unsigned*)(ar + arb + (size_t)(m0 + row) * 1024 + j0) + jp,
               (unsigned*)(sA[buf] + (row >> 4) * 512 + fragOff(row & 15, jp * 2)));
    }
    // B tiles: B[k=j][n=d]; pairs along j are strided -> pack two bf16.
#pragma unroll 4
    for (int e = tid; e < 4096; e += 256) {
      int d = e & 255, jp = e >> 8;
      const bhalf* s = src + srb + (size_t)(j0 + jp * 2) * 2048 + d0 + d;
      storePair(sB[buf] + (d >> 4) * 512, fragOff(d & 15, jp * 2),
                pack2bf(s[0], s[2048]));
    }
    // Prefetch the slab after next (global_prefetch_b8).
    if (j0 + 32 < 1024) {
      const bhalf* pf =
          src + srb + (size_t)(j0 + 32 + (tid >> 3)) * 2048 + d0 + (tid & 7) * 32;
      __builtin_prefetch(pf, 0, 1);
    }
  };

  stage(0, 0);
  asyncFence();
  __syncthreads();

  for (int j0 = 0; j0 < 1024; j0 += 32) {
    int cur = (j0 >> 5) & 1;
    if (j0 + 32 < 1024) stage(cur ^ 1, j0 + 32);  // overlap with compute below
#pragma unroll
    for (int t = 0; t < 2; ++t) {
      v16bf bm = loadFrag(sB[cur] + (wv * 2 + t) * 512);
#pragma unroll
      for (int mt = 0; mt < 4; ++mt) {
        v16bf am = loadFrag(sA[cur] + mt * 512);
        acc[mt][t] = WMMA_BF16(am, bm, acc[mt][t]);
      }
    }
    asyncFence();
    __syncthreads();
  }

  int ln = tid & 31, n = ln & 15, mb = (ln >> 4) << 3;
#pragma unroll
  for (int t = 0; t < 2; ++t) {
    int dcol = d0 + (wv * 2 + t) * 16 + n;
#pragma unroll
    for (int mt = 0; mt < 4; ++mt)
#pragma unroll
      for (int r = 0; r < 8; ++r)
        dst[srb + (size_t)(m0 + mt * 16 + mb + r) * 2048 + dcol] =
            (bhalf)fmaxf(acc[mt][t][r], 0.f);
  }
}

// ---------------------------------------------------------------------------
// K5: per-(b,h,l): attn = q2 @ k2^T (K=32), o = attn @ v (K=64), scatter
// through the reference's scrambled reshape chain to (b,c,h,w) f32.
// ---------------------------------------------------------------------------
__global__ __launch_bounds__(128) void attn_kernel(
    const bhalf* __restrict__ q2, const bhalf* __restrict__ k2,
    const bhalf* __restrict__ v, float* __restrict__ out) {
  __shared__ __align__(32) bhalf sQ[4 * 512];  // A tiles (rows tt)
  __shared__ __align__(32) bhalf sK[4 * 512];  // B tiles (n=tt, k=cc)
  __shared__ __align__(32) bhalf sV[4 * 512];  // B tiles [kk(2)][nt(2)] (n=cc,k=tt)
  __shared__ __align__(32) bhalf sT[8 * 512];  // attn as A tiles [mt(4)][kk(2)]
  int l = blockIdx.x, bh = blockIdx.y;
  int b = bh >> 2, h = bh & 3;
  int tid = threadIdx.x, wv = tid >> 5;
  size_t tb = ((size_t)bh * 1024 + l) * 2048;
#pragma unroll 4
  for (int e = tid; e < 1024; e += 128) {
    {  // q2/k2: contiguous pairs along cc -> raw dword copies
      int row = e >> 4, ccp = e & 15;
      int off = fragOff(row & 15, ccp * 2);
      stageU32((const unsigned*)(q2 + tb + row * 32) + ccp,
               (unsigned*)(sQ + (row >> 4) * 512 + off));
      stageU32((const unsigned*)(k2 + tb + row * 32) + ccp,
               (unsigned*)(sK + (row >> 4) * 512 + off));
    }
    {  // v: B[k=tt][n=cc], pairs along tt strided -> pack two bf16
      int cc = e & 31, ttp = e >> 5;
      int tt = ttp * 2;
      const bhalf* vs = v + tb + tt * 32 + cc;
      storePair(sV + ((tt >> 5) * 2 + (cc >> 4)) * 512, fragOff(cc & 15, tt & 31),
                pack2bf(vs[0], vs[32]));
    }
  }
  asyncFence();
  __syncthreads();
  int ln = tid & 31, n = ln & 15, mb = (ln >> 4) << 3;

  // Phase 1: attn tiles; wave wv owns column strip nt = wv. Results written
  // directly into A-fragment order for phase 2.
  {
    v16bf bm = loadFrag(sK + wv * 512);
#pragma unroll
    for (int mt = 0; mt < 4; ++mt) {
      v16bf am = loadFrag(sQ + mt * 512);
      v8f acc = zero8();
      acc = WMMA_BF16(am, bm, acc);
      bhalf* tile = sT + (mt * 2 + (wv >> 1)) * 512;
      int kcol = ((wv & 1) << 4) + n;
#pragma unroll
      for (int r = 0; r < 8; ++r)
        tile[fragOff(mb + r, kcol)] = (bhalf)acc[r];
    }
  }
  __syncthreads();

  // Phase 2: o = attn @ v; wave wv owns row strip mt = wv; K = 64 (2 steps).
  v8f acc2[2];
  acc2[0] = zero8();
  acc2[1] = zero8();
#pragma unroll
  for (int kk = 0; kk < 2; ++kk) {
    v16bf am = loadFrag(sT + (wv * 2 + kk) * 512);
#pragma unroll
    for (int nt = 0; nt < 2; ++nt) {
      v16bf bm = loadFrag(sV + (kk * 2 + nt) * 512);
      acc2[nt] = WMMA_BF16(am, bm, acc2[nt]);
    }
  }

  // Scatter through the reference's reshape chain:
  //  L2 = tt*16 + l/64 ; c2 = h*32+cc ; ch = 2*(l%64) + c2/64 ;
  //  y = (L2/32)*8 + (c2%64)/8 ; x = (L2%32)*8 + (c2%64)%8
  int l_hi = l >> 6, l_lo = l & 63;
#pragma unroll
  for (int nt = 0; nt < 2; ++nt) {
#pragma unroll
    for (int r = 0; r < 8; ++r) {
      int tt = wv * 16 + mb + r;
      int cc = nt * 16 + n;
      int c2 = h * 32 + cc;
      int L2 = tt * 16 + l_hi;
      int ch = 2 * l_lo + (c2 >> 6);
      int c2m = c2 & 63;
      int y = ((L2 >> 5) << 3) + (c2m >> 3);
      int xx = ((L2 & 31) << 3) + (c2m & 7);
      out[(((size_t)b * 128 + ch) * 256 + y) * 256 + xx] = acc2[nt][r];
    }
  }
}

// ---------------------------------------------------------------------------
extern "C" void kernel_launch(void* const* d_in, const int* in_sizes, int n_in,
                              void* d_out, int out_size, void* d_ws, size_t ws_size,
                              hipStream_t stream) {
  (void)in_sizes; (void)n_in; (void)out_size; (void)ws_size;
  const float* x    = (const float*)d_in[0];
  const float* W    = (const float*)d_in[1];
  const float* bias = (const float*)d_in[2];
  float* out = (float*)d_out;

  char* ws = (char*)d_ws;
  const size_t QKV = (size_t)4 * 4 * 1024 * 64 * 32;  // 33,554,432 elems
  bhalf* q  = (bhalf*)ws; ws += QKV * 2;
  bhalf* k  = (bhalf*)ws; ws += QKV * 2;
  bhalf* v  = (bhalf*)ws; ws += QKV * 2;
  bhalf* q2 = (bhalf*)ws; ws += QKV * 2;
  bhalf* k2 = (bhalf*)ws; ws += QKV * 2;
  bhalf* qr = (bhalf*)ws; ws += (size_t)16 * 1024 * 32 * 2;
  bhalf* kr = (bhalf*)ws; ws += (size_t)16 * 1024 * 32 * 2;
  bhalf* ar = (bhalf*)ws;  // 16 * 1024 * 1024 elems

  qkv_kernel<<<dim3(1024, 4), 256, 0, stream>>>(x, W, bias, q, k, v);
  mean_kernel<<<dim3(1024, 16), 64, 0, stream>>>(q, k, qr, kr);
  ar_kernel<<<dim3(16, 16, 16), 256, 0, stream>>>(qr, kr, ar);
  apply_ar_kernel<<<dim3(8, 16, 16), 256, 0, stream>>>(ar, q, q2);
  apply_ar_kernel<<<dim3(8, 16, 16), 256, 0, stream>>>(ar, k, k2);
  attn_kernel<<<dim3(1024, 16), 128, 0, stream>>>(q2, k2, v, out);
}
